// Uniformity_11914239279699
// MI455X (gfx1250) — matrix operand order repaired and checked by reading, not compiled, not run on hardware
//
#include <hip/hip_runtime.h>
#include <math.h>

typedef __attribute__((ext_vector_type(16))) _Float16 v16h;
typedef __attribute__((ext_vector_type(8)))  _Float16 v8h;
typedef __attribute__((ext_vector_type(8)))  float    v8f;
typedef __attribute__((ext_vector_type(4)))  unsigned int v4u;
typedef __attribute__((ext_vector_type(8)))  int v8i;
typedef __attribute__((ext_vector_type(4)))  int v4i;

#define D_FEAT 128
#define TAU_F  2.0f

// LDS byte offset of a __shared__ object: addrspacecast generic->local, then
// ptrtoint (AS3 pointers are 32-bit offsets from the wave's LDS base).
typedef __attribute__((address_space(3))) const void lds_cv;
static __device__ __forceinline__ unsigned lds_offset(const void* p) {
  return (unsigned)(unsigned long long)(lds_cv*)p;
}

// ---------------------------------------------------------------------------
// Kernel 1: L2-normalize each row of z (f32) and store as f16 row-major.
// ---------------------------------------------------------------------------
__global__ void __launch_bounds__(128)
normalize_rows(const float* __restrict__ z, _Float16* __restrict__ zn) {
  __shared__ float red[D_FEAT];
  const int row = blockIdx.x;
  const int t   = threadIdx.x;
  const float v = z[row * D_FEAT + t];
  red[t] = v * v;
  __syncthreads();
#pragma unroll
  for (int s = 64; s > 0; s >>= 1) {
    if (t < s) red[t] += red[t + s];
    __syncthreads();
  }
  const float inv = rsqrtf(red[0]);
  zn[row * D_FEAT + t] = (_Float16)(v * inv);
}

// ---------------------------------------------------------------------------
// Build D# group1 for a 2-D f16 tile load: tensor is [8192][128] f16
// row-major, tile is [tileRows][128].
//   w0: data_size=1 (2 bytes) at bits[17:16]
//   w1: tensor_dim0[15:0]  << 16           (tensor_dim0 = 128)
//   w2: tensor_dim1[15:0]  << 16           (tensor_dim1 = 8192)
//   w3: tensor_dim1[31:16] | tile_dim0<<16 (tile_dim0 = 128)
//   w4: tile_dim1 (rows)   | tile_dim2<<16 (0)
//   w5: tensor_dim0_stride[31:0] = 128
//   w6,w7: upper stride bits / dim1 stride = 0
// ---------------------------------------------------------------------------
static __device__ __forceinline__ v8i make_g1(int tileRows) {
  v8i g;
  g[0] = 0x00010000;
  g[1] = (int)(128u << 16);
  g[2] = (int)(8192u << 16);
  g[3] = (int)(128u << 16);
  g[4] = tileRows;
  g[5] = 128;
  g[6] = 0;
  g[7] = 0;
  return g;
}

static __device__ __forceinline__ v4u make_g0(unsigned lds_addr,
                                              unsigned long long gaddr) {
  v4u g;
  g[0] = 1u;                                   // count=1, user descriptor
  g[1] = lds_addr;                             // LDS byte address
  g[2] = (unsigned)gaddr;                      // global_addr[31:0]
  g[3] = ((unsigned)(gaddr >> 32) & 0x01FFFFFFu) | 0x80000000u; // addr[56:32] | type=2
  return g;
}

// ---------------------------------------------------------------------------
// Kernel 2: TDM-staged WMMA Gram tile fused with exp(-tau*sq) + reduction.
// Workgroup = 256 threads = 8 waves; output tile 128 rows x 64 cols.
// Wave 0 DMAs the A panel (128x128 f16, 32KB) and B panel (64x128, 16KB)
// into LDS with tensor_load_to_lds, waits TENSORcnt, then all waves run
// 16 v_wmma_f32_16x16x32_f16 each from LDS fragments.
// ---------------------------------------------------------------------------
__global__ void __launch_bounds__(256)
gram_exp_sum(const _Float16* __restrict__ zn, float* __restrict__ partials) {
  __shared__ _Float16 Ash[128 * D_FEAT];  // 32 KB
  __shared__ _Float16 Bsh[64 * D_FEAT];   // 16 KB
  __shared__ float red[256];

  const int rowPanel = blockIdx.y * 128;
  const int colPanel = blockIdx.x * 64;

  if (threadIdx.x < 32) {  // wave 0 issues the two tensor DMAs
    const unsigned long long base = (unsigned long long)(uintptr_t)zn;
    const unsigned long long gA = base + (unsigned long long)rowPanel * (D_FEAT * 2);
    const unsigned long long gB = base + (unsigned long long)colPanel * (D_FEAT * 2);
    const v4u g0A = make_g0(lds_offset(Ash), gA);
    const v4u g0B = make_g0(lds_offset(Bsh), gB);
    const v8i g1A = make_g1(128);
    const v8i g1B = make_g1(64);
    const v4i gz4 = {0, 0, 0, 0};
    const v8i gz8 = {0, 0, 0, 0, 0, 0, 0, 0};
    __builtin_amdgcn_tensor_load_to_lds(g0A, g1A, gz4, gz4, gz8, 0);
    __builtin_amdgcn_tensor_load_to_lds(g0B, g1B, gz4, gz4, gz8, 0);
    __builtin_amdgcn_s_wait_tensorcnt(0);
  }
  __syncthreads();
  asm volatile("" ::: "memory");  // LDS was written behind the compiler's back

  const int lane = threadIdx.x & 31;
  const int wave = threadIdx.x >> 5;                 // 0..7
  const int m    = lane & 15;
  const int hi   = lane >> 4;

  // Fragment layouts used here (16-bit, wave32; contiguous k, same for A and B):
  //   A: lane holds row M=lane&15; halves 0..15 = K kh..kh+15,
  //      kh=(lane>>4)*16.
  //   B: lane holds col N=lane&15; halves 0..15 = K (lane>>4)*16 ..+15.
  const _Float16* aRow = Ash + (wave * 16 + m) * D_FEAT;
  const int khA = hi * 16;
  const int khB = hi * 16;

  v8f c[4] = {v8f{}, v8f{}, v8f{}, v8f{}};

#pragma unroll
  for (int t = 0; t < 4; ++t) {
    const int k0 = t * 32;
    v8h alo = *(const v8h*)(aRow + k0 + khA);
    v8h ahi = *(const v8h*)(aRow + k0 + 8 + khA);
    v16h a;
#pragma unroll
    for (int i = 0; i < 8; ++i) { a[i] = alo[i]; a[i + 8] = ahi[i]; }

#pragma unroll
    for (int ct = 0; ct < 4; ++ct) {
      const _Float16* bRow = Bsh + (ct * 16 + m) * D_FEAT;
      v16h b = *(const v16h*)(bRow + k0 + khB);
      c[ct] = __builtin_amdgcn_wmma_f32_16x16x32_f16(
          false, a, false, b, (short)0, c[ct], false, false);
    }
  }

  // Fused epilogue: w = exp(-tau * max(2 - 2g, 0)).
  float sum = 0.0f;
#pragma unroll
  for (int ct = 0; ct < 4; ++ct) {
#pragma unroll
    for (int r = 0; r < 8; ++r) {
      const float g  = c[ct][r];
      const float sq = fmaxf(2.0f - 2.0f * g, 0.0f);
      sum += __expf(-TAU_F * sq);
    }
  }

  red[threadIdx.x] = sum;
  __syncthreads();
#pragma unroll
  for (int s = 128; s > 0; s >>= 1) {
    if (threadIdx.x < s) red[threadIdx.x] += red[threadIdx.x + s];
    __syncthreads();
  }
  if (threadIdx.x == 0)
    partials[blockIdx.y * gridDim.x + blockIdx.x] = red[0];
}

// ---------------------------------------------------------------------------
// Kernel 3: reduce partials, apply triangle trick, compute loss.
// ---------------------------------------------------------------------------
__global__ void __launch_bounds__(256)
finalize_loss(const float* __restrict__ partials, int nparts,
              float* __restrict__ out, int n) {
  __shared__ float red[256];
  float s = 0.0f;
  for (int i = threadIdx.x; i < nparts; i += 256) s += partials[i];
  red[threadIdx.x] = s;
  __syncthreads();
#pragma unroll
  for (int st = 128; st > 0; st >>= 1) {
    if (threadIdx.x < st) red[threadIdx.x] += red[threadIdx.x + st];
    __syncthreads();
  }
  if (threadIdx.x == 0) {
    const float  S_full = red[0];
    const float  upper  = 0.5f * (S_full - (float)n);
    const double npairs = (double)n * (double)(n - 1) * 0.5;
    const float  gscore = (float)((double)upper / npairs);
    out[0] = logf(gscore) + 4.0f * TAU_F;
  }
}

// ---------------------------------------------------------------------------
extern "C" void kernel_launch(void* const* d_in, const int* in_sizes, int n_in,
                              void* d_out, int out_size, void* d_ws,
                              size_t ws_size, hipStream_t stream) {
  (void)n_in; (void)out_size; (void)ws_size;
  const float* z = (const float*)d_in[0];
  const int n = in_sizes[0] / D_FEAT;  // 8192

  _Float16* zn = (_Float16*)d_ws;
  float* partials =
      (float*)((char*)d_ws + (size_t)n * D_FEAT * sizeof(_Float16));

  normalize_rows<<<n, 128, 0, stream>>>(z, zn);

  dim3 grid(n / 64, n / 128);
  gram_exp_sum<<<grid, 256, 0, stream>>>(zn, partials);

  const int nparts = (n / 64) * (n / 128);
  finalize_loss<<<1, 256, 0, stream>>>(partials, nparts, (float*)d_out, n);
}
